// EDMBlock_85521388798350
// MI455X (gfx1250) — compile-verified, matrix operand order checked by reading, not run on hardware
//
#include <hip/hip_runtime.h>
#include <hip/hip_bf16.h>

// ---------------------------------------------------------------------------
// B=16, C=32 (BC=512), LOOKBACK=1024, PRED=256, TOTAL=1280, DELAY=64,
// DPD=128, UP=244, keys=971, flat=15616.  Heavy math: f16 WMMA
// (v_wmma_f32_16x16x32_f16) with fp32 accumulation; softmax/LN stats fp32.
//
// Fragment layout fact (ISA 7.12.2, 16-bit 16x32 A / 32x16 B): a lane's 16
// halfs are two contiguous K-groups: K in [8*hi, 8*hi+8) and
// [16+8*hi, 16+8*hi+8)  ->  each fragment = two 16-byte loads when the
// source is K-contiguous (row-major A, transposed B).
// ---------------------------------------------------------------------------

typedef __attribute__((ext_vector_type(16))) _Float16 v16h;
typedef __attribute__((ext_vector_type(8)))  _Float16 v8h;
typedef __attribute__((ext_vector_type(8)))  float    v8f;

#define BC_TOT   512
#define SLEN     1280
#define SPAD     1344       // series padded with zeros (branchless windows)
#define NKEY     971
#define KROWS    992        // keys padded to 62 tiles (even # of 16-tiles)
#define QROWS    256
#define NQ       244
#define DPD      128
#define DELAY    64
#define FLATN    15616
#define NOUT     256

// two aligned 16B loads -> one v16h fragment (f16 source, K-contiguous)
__device__ __forceinline__ v16h load_frag_h16(const _Float16* base) {
    v8h lo = *(const v8h*)(base);
    v8h hi = *(const v8h*)(base + 16);
    return __builtin_shufflevector(lo, hi, 0, 1, 2, 3, 4, 5, 6, 7,
                                           8, 9, 10, 11, 12, 13, 14, 15);
}

// fragment from fp32 source, unaligned (scalar b32 loads + cvt)
__device__ __forceinline__ v16h frag_from_f32(const float* p0) {
    v16h r;
#pragma unroll
    for (int e = 0; e < 8; ++e) {
        r[e]     = (_Float16)p0[e];
        r[8 + e] = (_Float16)p0[16 + e];
    }
    return r;
}

// ---------------------------------------------------------------------------
// Kernel 0a: out[n*K + k] = (f16) in[k*N + n]   (weight transpose + cvt)
// ---------------------------------------------------------------------------
__global__ void transpose_cvt_kernel(const float* __restrict__ in,
                                     _Float16* __restrict__ out, int K, int N) {
    int idx = blockIdx.x * blockDim.x + threadIdx.x;
    if (idx < K * N) {
        int k = idx % K;         // k fastest -> contiguous writes
        int n = idx / K;
        out[(size_t)n * K + k] = (_Float16)in[(size_t)k * N + n];
    }
}

// ---------------------------------------------------------------------------
// Kernel 0b: VT[bc][d][key] = (f16) series[key + 1 + d]  (td_next, transposed
// so PV B-fragments are key-contiguous -> 2 x b128 loads).  key < 992 padded;
// max series index = 991 + 1 + 63 = 1055 < 1280, always in range.
// ---------------------------------------------------------------------------
__global__ void vt_prep_kernel(const float* __restrict__ lookback,
                               const float* __restrict__ focal,
                               _Float16* __restrict__ VT) {   // [512][64][992]
    const int bc  = blockIdx.y;
    const int idx = blockIdx.x * blockDim.x + threadIdx.x;    // 0..63487
    const int d   = idx / KROWS;
    const int key = idx - d * KROWS;
    const int i   = key + 1 + d;
    float v = (i < 1024) ? lookback[(size_t)bc * 1024 + i]
                         : focal[(size_t)bc * 256 + (i - 1024)];
    VT[(size_t)bc * 64 * KROWS + idx] = (_Float16)v;
}

// ---------------------------------------------------------------------------
// Kernel 1: Hankel-window projection + bias + PE + LayerNorm -> Kf16/Qf16.
// One wave per block; tiles 0..61 -> K (td_hist), 62..77 -> Q (focal).
// ---------------------------------------------------------------------------
__global__ __launch_bounds__(32) void proj_ln_kernel(
    const float* __restrict__ lookback,   // [512][1024]
    const float* __restrict__ focal,      // [512][256]
    const float* __restrict__ Wp,         // [64][128]
    const float* __restrict__ bp,         // [128]
    const float* __restrict__ pe,         // [1280][128]
    const float* __restrict__ ln_g,       // [128]
    const float* __restrict__ ln_b,       // [128]
    _Float16* __restrict__ Kf,            // [512][992][128]
    _Float16* __restrict__ Qf)            // [512][256][128]
{
    __shared__ __align__(16) float    s_series[SPAD];
    __shared__ __align__(16) _Float16 s_wt[DPD * DELAY];  // transposed: [col][64]
    __shared__ __align__(16) float    s_proj[16 * DPD];

    const int lane = threadIdx.x;
    const int bc   = blockIdx.z * 32 + blockIdx.y;
    const int t    = blockIdx.x;

    for (int i = lane; i < SPAD; i += 32)
        s_series[i] = (i < 1024) ? lookback[(size_t)bc * 1024 + i]
                    : (i < SLEN) ? focal[(size_t)bc * 256 + (i - 1024)]
                                 : 0.0f;
    // W_proj transposed into LDS as f16: s_wt[col*64 + k] = Wp[k][col]
    for (int i = lane; i < DPD * DELAY; i += 32) {
        int col = i >> 6, k = i & 63;
        s_wt[i] = (_Float16)Wp[(size_t)k * DPD + col];
    }
    __syncthreads();

    const bool isK   = (t < 62);
    const int  r0    = isK ? t * 16 : (t - 62) * 16;
    const int  u0    = isK ? 0 : 972;
    const int  peoff = isK ? 0 : NKEY;

    const int rowA = lane & 15;
    const int hi   = lane >> 4;

    // A (Hankel): A[row][k] = series[u0 + r0 + row + k]; padded -> branchless
    v16h afrag[2];
#pragma unroll
    for (int ks = 0; ks < 2; ++ks)
        afrag[ks] = frag_from_f32(s_series + u0 + r0 + rowA + ks * 32 + 8 * hi);

    for (int nt = 0; nt < 8; ++nt) {
        const int col = nt * 16 + rowA;
        v8f acc = {};
#pragma unroll
        for (int ks = 0; ks < 2; ++ks) {
            v16h bfrag = load_frag_h16(s_wt + col * DELAY + ks * 32 + 8 * hi);
            acc = __builtin_amdgcn_wmma_f32_16x16x32_f16(
                false, afrag[ks], false, bfrag, (short)0, acc, false, false);
        }
#pragma unroll
        for (int v = 0; v < 8; ++v)
            s_proj[(v + 8 * hi) * DPD + col] = acc[v];
    }
    __syncthreads();

    // LayerNorm: lane (row = lane&15) handles 64 cols; halves merged via shfl
    {
        const int row = lane & 15;
        const int c0  = hi * 64;
        const int pr  = peoff + r0 + row;
        float s1 = 0.f, s2 = 0.f;
        for (int i = 0; i < 64; ++i) {
            int   col = c0 + i;
            float x = s_proj[row * DPD + col] + bp[col] + pe[(size_t)pr * DPD + col];
            s1 += x; s2 += x * x;
        }
        float sum  = s1 + __shfl_xor(s1, 16);
        float sum2 = s2 + __shfl_xor(s2, 16);
        float mean = sum * (1.0f / 128.0f);
        float var  = sum2 * (1.0f / 128.0f) - mean * mean;
        float inv  = rsqrtf(var + 1e-5f);
        _Float16* dst = isK
            ? (Kf + (size_t)bc * KROWS * DPD + (size_t)(r0 + row) * DPD)
            : (Qf + (size_t)bc * QROWS * DPD + (size_t)(r0 + row) * DPD);
        for (int i = 0; i < 64; ++i) {
            int   col = c0 + i;
            float x = s_proj[row * DPD + col] + bp[col] + pe[(size_t)pr * DPD + col];
            dst[col] = (_Float16)((x - mean) * inv * ln_g[col] + ln_b[col]);
        }
    }
}

// ---------------------------------------------------------------------------
// Kernel 2: flash attention.  One wave per (bc, 16-q-row) tile; stream 32
// keys/iter; online softmax fp32; P re-fragmented through 1KB LDS;
// V fragments are 2 x b128 loads from pre-transposed f16 VT.
// ---------------------------------------------------------------------------
__global__ __launch_bounds__(32) void attn_kernel(
    const _Float16* __restrict__ Kf,      // [512][992][128]
    const _Float16* __restrict__ Qf,      // [512][256][128]
    const _Float16* __restrict__ VT,      // [512][64][992]
    _Float16*       __restrict__ preds)   // [512][15616] (f16)
{
    __shared__ __align__(16) _Float16 s_p[16 * 32];

    const int lane = threadIdx.x;
    const int bc   = blockIdx.z * 32 + blockIdx.y;
    const int qt   = blockIdx.x;
    const int rowA = lane & 15;
    const int hi   = lane >> 4;

    const _Float16* Qb  = Qf + (size_t)bc * QROWS * DPD + (size_t)qt * 16 * DPD;
    const _Float16* Kb  = Kf + (size_t)bc * KROWS * DPD;
    const _Float16* VTb = VT + (size_t)bc * 64 * KROWS;

    // resident Q fragments: 2 x b128 loads each
    v16h qfrag[4];
#pragma unroll
    for (int ks = 0; ks < 4; ++ks)
        qfrag[ks] = load_frag_h16(Qb + rowA * DPD + ks * 32 + 8 * hi);

    v8f   o[4] = {{}, {}, {}, {}};
    float mrow[8], lrow[8];
#pragma unroll
    for (int v = 0; v < 8; ++v) { mrow[v] = -1e30f; lrow[v] = 0.0f; }

    const float scale = 0.08838834764831845f;   // 1/sqrt(128)

    for (int kp = 0; kp < 31; ++kp) {           // 31*32 = 992 padded keys
        if (kp + 1 < 31) {                      // cover latency of next tile
            __builtin_prefetch(Kb + (size_t)((kp + 1) * 32 + lane) * DPD, 0, 3);
            __builtin_prefetch(VTb + (size_t)lane * KROWS + (kp + 1) * 32, 0, 3);
        }
        v8f s01[2];
#pragma unroll
        for (int h = 0; h < 2; ++h) {
            const int key0 = kp * 32 + h * 16;
            const _Float16* kr = Kb + (size_t)(key0 + rowA) * DPD;
            v8f sacc = {};
#pragma unroll
            for (int ks = 0; ks < 4; ++ks) {
                v16h bfrag = load_frag_h16(kr + ks * 32 + 8 * hi);  // B = K^T
                sacc = __builtin_amdgcn_wmma_f32_16x16x32_f16(
                    false, qfrag[ks], false, bfrag, (short)0, sacc, false, false);
            }
            const int keycol = key0 + rowA;
#pragma unroll
            for (int v = 0; v < 8; ++v)
                sacc[v] = (keycol < NKEY) ? sacc[v] * scale : -1e30f;
            s01[h] = sacc;
        }

        float corr[8];
#pragma unroll
        for (int v = 0; v < 8; ++v) {
            float rm = fmaxf(s01[0][v], s01[1][v]);
            rm = fmaxf(rm, __shfl_xor(rm, 1));
            rm = fmaxf(rm, __shfl_xor(rm, 2));
            rm = fmaxf(rm, __shfl_xor(rm, 4));
            rm = fmaxf(rm, __shfl_xor(rm, 8));
            float mnew = fmaxf(mrow[v], rm);
            float c  = __expf(mrow[v] - mnew);
            float p0 = __expf(s01[0][v] - mnew);
            float p1 = __expf(s01[1][v] - mnew);
            float rs = p0 + p1;
            rs += __shfl_xor(rs, 1);
            rs += __shfl_xor(rs, 2);
            rs += __shfl_xor(rs, 4);
            rs += __shfl_xor(rs, 8);
            lrow[v] = lrow[v] * c + rs;
            mrow[v] = mnew;
            corr[v] = c;
            const int prow = v + 8 * hi;
            s_p[prow * 32 + rowA]      = (_Float16)p0;
            s_p[prow * 32 + rowA + 16] = (_Float16)p1;
        }
#pragma unroll
        for (int nt = 0; nt < 4; ++nt)
#pragma unroll
            for (int v = 0; v < 8; ++v)
                o[nt][v] *= corr[v];
        __syncthreads();

        // P (16x32) as A fragment: 2 x ds_load_b128
        v16h pfrag = load_frag_h16(s_p + rowA * 32 + 8 * hi);
        // V fragments from VT (key-contiguous): 2 x global b128 each
#pragma unroll
        for (int nt = 0; nt < 4; ++nt) {
            const int d = nt * 16 + rowA;
            v16h vfrag = load_frag_h16(VTb + (size_t)d * KROWS + kp * 32 + 8 * hi);
            o[nt] = __builtin_amdgcn_wmma_f32_16x16x32_f16(
                false, pfrag, false, vfrag, (short)0, o[nt], false, false);
        }
        __syncthreads();
    }

#pragma unroll
    for (int nt = 0; nt < 4; ++nt)
#pragma unroll
        for (int v = 0; v < 8; ++v) {
            const int rowg = qt * 16 + v + 8 * hi;
            if (rowg < NQ)
                preds[(size_t)bc * FLATN + (size_t)rowg * DELAY + nt * 16 + rowA] =
                    (_Float16)(o[nt][v] / lrow[v]);
        }
}

// ---------------------------------------------------------------------------
// Kernel 3: h = gelu(flat @ W1 + b1).  M=512, K=15616, N=256.
// flat (preds) is f16; W1T is pre-transposed f16 -> all frags 2 x b128.
// ---------------------------------------------------------------------------
__global__ __launch_bounds__(32) void mlp1_kernel(
    const _Float16* __restrict__ flat,   // [512][15616] f16
    const _Float16* __restrict__ W1T,    // [256][15616]
    const float*    __restrict__ b1,     // [256]
    _Float16*       __restrict__ h)      // [512][256]
{
    const int lane = threadIdx.x;
    const int rowA = lane & 15;
    const int hi   = lane >> 4;
    const int mt   = blockIdx.x;      // 0..31
    const int nq   = blockIdx.y;      // 0..7

    v8f acc[2] = {{}, {}};
    const _Float16* arow = flat + (size_t)(mt * 16 + rowA) * FLATN;
    const _Float16* b0   = W1T + (size_t)(nq * 32 + rowA) * FLATN;
    const _Float16* b1r  = W1T + (size_t)(nq * 32 + 16 + rowA) * FLATN;

    for (int kst = 0; kst < FLATN / 32; ++kst) {
        const int kb = kst * 32 + 8 * hi;
        v16h afrag = load_frag_h16(arow + kb);
        v16h bf0   = load_frag_h16(b0 + kb);
        v16h bf1   = load_frag_h16(b1r + kb);
        acc[0] = __builtin_amdgcn_wmma_f32_16x16x32_f16(
            false, afrag, false, bf0, (short)0, acc[0], false, false);
        acc[1] = __builtin_amdgcn_wmma_f32_16x16x32_f16(
            false, afrag, false, bf1, (short)0, acc[1], false, false);
    }
#pragma unroll
    for (int nt = 0; nt < 2; ++nt) {
        const int col = nq * 32 + nt * 16 + rowA;
#pragma unroll
        for (int v = 0; v < 8; ++v) {
            const int row = mt * 16 + v + 8 * hi;
            float x = acc[nt][v] + b1[col];
            float g = 0.5f * x * (1.0f + erff(x * 0.7071067811865475f)); // exact gelu
            h[(size_t)row * NOUT + col] = (_Float16)g;
        }
    }
}

// ---------------------------------------------------------------------------
// Kernel 4: out = h @ W2 + b2.  M=512, K=256, N=256.  W2T f16 [256][256].
// ---------------------------------------------------------------------------
__global__ __launch_bounds__(32) void mlp2_kernel(
    const _Float16* __restrict__ h,    // [512][256]
    const _Float16* __restrict__ W2T,  // [256][256]
    const float*    __restrict__ b2,   // [256]
    float*          __restrict__ out)  // [512][256]
{
    const int lane = threadIdx.x;
    const int rowA = lane & 15;
    const int hi   = lane >> 4;
    const int mt   = blockIdx.x;      // 0..31
    const int nq   = blockIdx.y;      // 0..7

    v8f acc[2] = {{}, {}};
    const _Float16* arow = h + (size_t)(mt * 16 + rowA) * NOUT;
    const _Float16* b0   = W2T + (size_t)(nq * 32 + rowA) * NOUT;
    const _Float16* b1r  = W2T + (size_t)(nq * 32 + 16 + rowA) * NOUT;

#pragma unroll
    for (int kst = 0; kst < NOUT / 32; ++kst) {
        const int kb = kst * 32 + 8 * hi;
        v16h afrag = load_frag_h16(arow + kb);
        v16h bf0   = load_frag_h16(b0 + kb);
        v16h bf1   = load_frag_h16(b1r + kb);
        acc[0] = __builtin_amdgcn_wmma_f32_16x16x32_f16(
            false, afrag, false, bf0, (short)0, acc[0], false, false);
        acc[1] = __builtin_amdgcn_wmma_f32_16x16x32_f16(
            false, afrag, false, bf1, (short)0, acc[1], false, false);
    }
#pragma unroll
    for (int nt = 0; nt < 2; ++nt) {
        const int col = nq * 32 + nt * 16 + rowA;
#pragma unroll
        for (int v = 0; v < 8; ++v) {
            const int row = mt * 16 + v + 8 * hi;
            out[(size_t)row * NOUT + col] = acc[nt][v] + b2[col];
        }
    }
}

// ---------------------------------------------------------------------------
extern "C" void kernel_launch(void* const* d_in, const int* in_sizes, int n_in,
                              void* d_out, int out_size, void* d_ws, size_t ws_size,
                              hipStream_t stream) {
    (void)in_sizes; (void)n_in; (void)out_size; (void)ws_size;

    const float* lookback = (const float*)d_in[0];
    const float* focal    = (const float*)d_in[1];
    const float* Wp       = (const float*)d_in[2];
    const float* bp       = (const float*)d_in[3];
    const float* pe       = (const float*)d_in[4];
    const float* ln_g     = (const float*)d_in[5];
    const float* ln_b     = (const float*)d_in[6];
    const float* W1       = (const float*)d_in[7];
    const float* b1       = (const float*)d_in[8];
    const float* W2       = (const float*)d_in[9];
    const float* b2       = (const float*)d_in[10];
    float* out = (float*)d_out;

    char*  ws  = (char*)d_ws;
    size_t off = 0;
    _Float16* Kf    = (_Float16*)(ws + off); off += (size_t)BC_TOT * KROWS * DPD * sizeof(_Float16);
    _Float16* Qf    = (_Float16*)(ws + off); off += (size_t)BC_TOT * QROWS * DPD * sizeof(_Float16);
    _Float16* preds = (_Float16*)(ws + off); off += (size_t)BC_TOT * FLATN * sizeof(_Float16);
    _Float16* hb    = (_Float16*)(ws + off); off += (size_t)BC_TOT * NOUT * sizeof(_Float16);
    _Float16* W1T   = (_Float16*)(ws + off); off += (size_t)NOUT * FLATN * sizeof(_Float16);
    _Float16* W2T   = (_Float16*)(ws + off); off += (size_t)NOUT * NOUT * sizeof(_Float16);
    _Float16* VT    = (_Float16*)(ws + off); off += (size_t)BC_TOT * 64 * KROWS * sizeof(_Float16);

    // 0) weight transposes (f32 -> f16, [K][N] -> [N][K]) + VT (td_next^T f16)
    transpose_cvt_kernel<<<(FLATN * NOUT + 255) / 256, 256, 0, stream>>>(W1, W1T, FLATN, NOUT);
    transpose_cvt_kernel<<<(NOUT * NOUT + 255) / 256, 256, 0, stream>>>(W2, W2T, NOUT, NOUT);
    vt_prep_kernel<<<dim3((64 * KROWS) / 256, BC_TOT), 256, 0, stream>>>(lookback, focal, VT);
    // 1) projections + PE + layernorm -> Kf16 / Qf16
    proj_ln_kernel<<<dim3(78, 32, 16), 32, 0, stream>>>(
        lookback, focal, Wp, bp, pe, ln_g, ln_b, Kf, Qf);
    // 2) flash attention -> preds (flat, f16)
    attn_kernel<<<dim3(16, 32, 16), 32, 0, stream>>>(Kf, Qf, VT, preds);
    // 3) hidden = gelu(flat @ W1 + b1)
    mlp1_kernel<<<dim3(32, 8), 32, 0, stream>>>(preds, W1T, b1, hb);
    // 4) out = hidden @ W2 + b2
    mlp2_kernel<<<dim3(32, 8), 32, 0, stream>>>(hb, W2T, b2, out);
}
